// WindowAttention_55250459295902
// MI455X (gfx1250) — compile-verified, matrix operand order checked by reading, not run on hardware
//
#include <hip/hip_runtime.h>

typedef __attribute__((ext_vector_type(16))) _Float16 v16h;
typedef __attribute__((ext_vector_type(8)))  float    v8f;
typedef __attribute__((ext_vector_type(4)))  unsigned int v4u;

#define LDX 200   // row stride (halves) for X/Q/K/O buffers: 100 dwords -> conflict-free 16-lane row access
#define LDP 72    // row stride (halves) for P and Vt: 36 dwords -> conflict-free

static __device__ __forceinline__ v8f wmma16(v16h a, v16h b, v8f c) {
  return __builtin_amdgcn_wmma_f32_16x16x32_f16(false, a, false, b, (short)0, c, false, false);
}

// 16x32 f16 A-fragment (or 32x16 B-fragment, same per-lane pattern) from an
// f16 row-major matrix: lane l holds row/col (r0+l%16), halves 0-7 at
// c0+(l>=16?8:0), halves 8-15 at +16.  Two 16B loads per lane.
static __device__ __forceinline__ v16h ld_frag(const _Float16* __restrict__ base,
                                               int r0, int c0, int ld, int lane) {
  const int lr  = lane & 15;
  const int off = (lane & 16) ? 8 : 0;
  const _Float16* p = base + (r0 + lr) * ld + c0 + off;
  union { v4u u[2]; v16h h; } f;
  f.u[0] = *(const v4u*)(p);
  f.u[1] = *(const v4u*)(p + 16);
  return f.h;
}

// Same fragment but sourced from an f32 row-major matrix (convert inline).
static __device__ __forceinline__ v16h ld_frag_f32(const float* __restrict__ W,
                                                   int r0, int c0, int ld, int lane) {
  const int lr  = lane & 15;
  const int off = (lane & 16) ? 8 : 0;
  const float* p = W + (size_t)(r0 + lr) * ld + c0 + off;
  const float4* q = (const float4*)p;
  float4 f0 = q[0], f1 = q[1], f2 = q[4], f3 = q[5];
  union { _Float16 e[16]; v16h h; } f;
  f.e[0]  = (_Float16)f0.x; f.e[1]  = (_Float16)f0.y; f.e[2]  = (_Float16)f0.z; f.e[3]  = (_Float16)f0.w;
  f.e[4]  = (_Float16)f1.x; f.e[5]  = (_Float16)f1.y; f.e[6]  = (_Float16)f1.z; f.e[7]  = (_Float16)f1.w;
  f.e[8]  = (_Float16)f2.x; f.e[9]  = (_Float16)f2.y; f.e[10] = (_Float16)f2.z; f.e[11] = (_Float16)f2.w;
  f.e[12] = (_Float16)f3.x; f.e[13] = (_Float16)f3.y; f.e[14] = (_Float16)f3.z; f.e[15] = (_Float16)f3.w;
  return f.h;
}

// Convert weight matrices to f16 once per launch (they stay L2-resident).
__global__ void winattn_cvt_weights(const float* __restrict__ qkv_w,
                                    const float* __restrict__ proj_w,
                                    _Float16* __restrict__ dst) {
  int i = blockIdx.x * 256 + threadIdx.x;
  if (i < 576 * 192) dst[i] = (_Float16)qkv_w[i];
  if (i < 192 * 192) dst[576 * 192 + i] = (_Float16)proj_w[i];
}

// One workgroup = one 7x7 window.  192 threads = 6 wave32 (one head per wave
// in the attention phase).  launch_bounds min-waves keeps VGPRs <= 256 so we
// avoid s_set_vgpr_msb juggling and keep occupancy at the LDS limit.
template <bool W16>
__global__ __launch_bounds__(192, 2) void winattn_kernel(
    const float* __restrict__ x,          // [Bnw,49,192]
    const float* __restrict__ mask,       // [nw,49,49]
    int nw,
    const float* __restrict__ qkv_w,      // [576,192] f32
    const _Float16* __restrict__ qkv_wh,  // [576,192] f16 (ws) or null
    const float* __restrict__ qkv_b,      // [576]
    const float* __restrict__ proj_w,     // [192,192] f32
    const _Float16* __restrict__ proj_wh, // [192,192] f16 (ws) or null
    const float* __restrict__ proj_b,     // [192]
    const float* __restrict__ bias_table, // [169,6]
    float* __restrict__ out)              // [Bnw,49,192]
{
  __shared__ __align__(16) _Float16 Xh[64 * LDX];        // X (f16), reused as O
  __shared__ __align__(16) _Float16 Qh[64 * LDX];
  __shared__ __align__(16) _Float16 Kh[64 * LDX];
  __shared__ __align__(16) _Float16 Vt[192 * LDP];       // V transposed: [d_global][token]
  __shared__ __align__(16) _Float16 Pw[6 * 64 * LDP];    // per-wave softmax probs

  const int tid  = threadIdx.x;
  const int w    = blockIdx.x;
  const int wv   = tid >> 5;
  const int lane = tid & 31;
  const int lr   = lane & 15;
  const int rsel = (lane & 16) ? 8 : 0;

  // ---- Phase 0: zero pad rows of Xh, load X f32 -> f16 ----
  {
    unsigned int* z = (unsigned int*)(Xh + 49 * LDX);
    for (int i = tid; i < 15 * LDX / 2; i += 192) z[i] = 0u;
    const float4* xs = (const float4*)(x + (size_t)w * 9408);
    for (int i = tid; i < 2352; i += 192) {           // 49 rows * 48 float4
      float4 v = xs[i];
      int r = i / 48, c = (i % 48) * 4;
      _Float16* d = Xh + r * LDX + c;
      d[0] = (_Float16)v.x; d[1] = (_Float16)v.y;
      d[2] = (_Float16)v.z; d[3] = (_Float16)v.w;
    }
  }
  __syncthreads();

  // ---- Phase A: QKV = X @ qkv_w^T + qkv_b ----
  // Each wave: all 4 M-blocks x 6 N-tiles (nt = wv + 6j).  A-fragments for a
  // M-block are loaded once into registers and reused for all 6 N-tiles.
  // Outer loops kept rolled to bound register pressure.
#pragma clang loop unroll(disable)
  for (int mt = 0; mt < 4; ++mt) {
    const int m0 = mt * 16;
    const int rb = m0 + rsel;
    v16h a[6];
#pragma unroll
    for (int kc = 0; kc < 6; ++kc) a[kc] = ld_frag(Xh, m0, kc * 32, LDX, lane);
#pragma clang loop unroll(disable)
    for (int j = 0; j < 6; ++j) {
      const int n0 = (wv + 6 * j) * 16;
      v8f acc = {0.f, 0.f, 0.f, 0.f, 0.f, 0.f, 0.f, 0.f};
#pragma unroll
      for (int kc = 0; kc < 6; ++kc) {
        v16h b = W16 ? ld_frag(qkv_wh, n0, kc * 32, 192, lane)
                     : ld_frag_f32(qkv_w, n0, kc * 32, 192, lane);
        acc = wmma16(a[kc], b, acc);
      }
      const int col = n0 + lr;
      const float bb = qkv_b[col];
      // whole 16-wide tile falls in exactly one of Q/K/V: branch is
      // wave-uniform -> force a scalar branch (no exec juggling).
      const int n0s = __builtin_amdgcn_readfirstlane(n0);
      if (n0s < 192) {
#pragma unroll
        for (int r = 0; r < 8; ++r)
          Qh[(rb + r) * LDX + col] = (_Float16)(acc[r] + bb);
      } else if (n0s < 384) {
#pragma unroll
        for (int r = 0; r < 8; ++r)
          Kh[(rb + r) * LDX + (col - 192)] = (_Float16)(acc[r] + bb);
      } else {
        // transpose V: row = feature (col-384), column = token; adjacent r
        // are adjacent tokens -> pack pairs into 32-bit stores.
        _Float16* vrow = Vt + (col - 384) * LDP + rb;
#pragma unroll
        for (int rr = 0; rr < 4; ++rr) {
          union { _Float16 e[2]; unsigned int u; } pk;
          pk.e[0] = (_Float16)(acc[2 * rr] + bb);
          pk.e[1] = (_Float16)(acc[2 * rr + 1] + bb);
          *(unsigned int*)(vrow + 2 * rr) = pk.u;
        }
      }
    }
  }
  __syncthreads();

  // ---- Phase B: per-head attention (head h = wave id) ----
  {
    const int h = wv;                     // H == 6 == number of waves
    const float scale = 0.17677669529663687f;   // 32^-0.5
    const size_t mbase = (size_t)(w % nw) * 2401;
    _Float16* P = Pw + wv * (64 * LDP);

#pragma clang loop unroll(disable)
    for (int mt = 0; mt < 4; ++mt) {
      const int m0 = mt * 16;
      const int rb = m0 + rsel;
      v16h aq = ld_frag(Qh, m0, h * 32, LDX, lane);   // hd == 32 == one K chunk
      v8f s[4];
#pragma unroll
      for (int nt = 0; nt < 4; ++nt) {
        v16h bk = ld_frag(Kh, nt * 16, h * 32, LDX, lane);
        v8f z = {0.f, 0.f, 0.f, 0.f, 0.f, 0.f, 0.f, 0.f};
        s[nt] = wmma16(aq, bk, z);
      }
      // epilogue: scale + relative-position bias + mask, pad cols -> -1e30
#pragma unroll
      for (int r = 0; r < 8; ++r) {
        const int m  = rb + r;
        const int mq = (m < 49) ? m : 48;
        const int qy = mq / 7, qx = mq - qy * 7;
#pragma unroll
        for (int nt = 0; nt < 4; ++nt) {
          const int k = nt * 16 + lr;
          float v;
          if (k < 49) {
            const int ky  = k / 7, kx = k - ky * 7;
            const int idx = (qy - ky + 6) * 13 + (qx - kx + 6);
            v = s[nt][r] * scale + bias_table[idx * 6 + h]
                + mask[mbase + mq * 49 + k];
          } else {
            v = -1e30f;
          }
          s[nt][r] = v;
        }
      }
      // softmax per row (each row lives in one 16-lane half)
#pragma unroll
      for (int r = 0; r < 8; ++r) {
        float a0 = s[0][r], a1 = s[1][r], a2 = s[2][r], a3 = s[3][r];
        float mx = fmaxf(fmaxf(a0, a1), fmaxf(a2, a3));
#pragma unroll
        for (int d = 1; d < 16; d <<= 1) mx = fmaxf(mx, __shfl_xor(mx, d, 32));
        float e0 = __expf(a0 - mx), e1 = __expf(a1 - mx);
        float e2 = __expf(a2 - mx), e3 = __expf(a3 - mx);
        float sm = (e0 + e1) + (e2 + e3);
#pragma unroll
        for (int d = 1; d < 16; d <<= 1) sm += __shfl_xor(sm, d, 32);
        const float inv = 1.0f / sm;
        _Float16* pr = P + (rb + r) * LDP + lr;
        pr[0]  = (_Float16)(e0 * inv);
        pr[16] = (_Float16)(e1 * inv);
        pr[32] = (_Float16)(e2 * inv);
        pr[48] = (_Float16)(e3 * inv);
      }
      asm volatile("s_wait_dscnt 0" ::: "memory");  // P stores visible to own loads

      // O(mt block) = P @ V   (writes into Xh, reused as O buffer)
#pragma unroll
      for (int dt = 0; dt < 2; ++dt) {
        v8f o = {0.f, 0.f, 0.f, 0.f, 0.f, 0.f, 0.f, 0.f};
#pragma unroll
        for (int kc = 0; kc < 2; ++kc) {
          v16h ap = ld_frag(P, m0, kc * 32, LDP, lane);
          v16h bv = ld_frag(Vt, h * 32 + dt * 16, kc * 32, LDP, lane);
          o = wmma16(ap, bv, o);
        }
        const int cold = h * 32 + dt * 16 + lr;
#pragma unroll
        for (int r = 0; r < 8; ++r)
          Xh[(rb + r) * LDX + cold] = (_Float16)o[r];
      }
    }
  }
  __syncthreads();

  // ---- Phase C: out = O @ proj_w^T + proj_b ----
  // Each wave: 4 M-blocks x 2 N-tiles (nt = wv + 6j), A-fragments cached.
  float* orow = out + (size_t)w * 9408;
#pragma clang loop unroll(disable)
  for (int mt = 0; mt < 4; ++mt) {
    const int m0 = mt * 16;
    const int rb = m0 + rsel;
    v16h a[6];
#pragma unroll
    for (int kc = 0; kc < 6; ++kc) a[kc] = ld_frag(Xh, m0, kc * 32, LDX, lane);
#pragma clang loop unroll(disable)
    for (int j = 0; j < 2; ++j) {
      const int n0 = (wv + 6 * j) * 16;
      v8f acc = {0.f, 0.f, 0.f, 0.f, 0.f, 0.f, 0.f, 0.f};
#pragma unroll
      for (int kc = 0; kc < 6; ++kc) {
        v16h b = W16 ? ld_frag(proj_wh, n0, kc * 32, 192, lane)
                     : ld_frag_f32(proj_w, n0, kc * 32, 192, lane);
        acc = wmma16(a[kc], b, acc);
      }
      const int col = n0 + lr;
      const float pb = proj_b[col];
#pragma unroll
      for (int r = 0; r < 8; ++r) {
        const int m = rb + r;
        if (m < 49) orow[m * 192 + col] = acc[r] + pb;
      }
    }
  }
}

extern "C" void kernel_launch(void* const* d_in, const int* in_sizes, int n_in,
                              void* d_out, int out_size, void* d_ws, size_t ws_size,
                              hipStream_t stream) {
  const float* x          = (const float*)d_in[0];
  const float* mask       = (const float*)d_in[1];
  const float* qkv_w      = (const float*)d_in[2];
  const float* qkv_b      = (const float*)d_in[3];
  const float* proj_w     = (const float*)d_in[4];
  const float* proj_b     = (const float*)d_in[5];
  const float* bias_table = (const float*)d_in[6];
  float* out = (float*)d_out;

  int Bnw = in_sizes[0] / 9408;          // 49*192
  int nw  = in_sizes[1] / 2401;          // 49*49
  if (nw <= 0) nw = 1;
  if (Bnw <= 0) return;

  const size_t need = (size_t)(576 * 192 + 192 * 192) * sizeof(_Float16);
  if (d_ws != nullptr && ws_size >= need) {
    _Float16* wh = (_Float16*)d_ws;
    winattn_cvt_weights<<<(576 * 192 + 255) / 256, 256, 0, stream>>>(qkv_w, proj_w, wh);
    winattn_kernel<true><<<Bnw, 192, 0, stream>>>(
        x, mask, nw, qkv_w, wh, qkv_b, proj_w, wh + 576 * 192, proj_b,
        bias_table, out);
  } else {
    winattn_kernel<false><<<Bnw, 192, 0, stream>>>(
        x, mask, nw, qkv_w, nullptr, qkv_b, proj_w, nullptr, proj_b,
        bias_table, out);
  }
}
